// VectorQuantizerEMA_4844723110233
// MI455X (gfx1250) — matrix operand order repaired and checked by reading, not compiled, not run on hardware
//
#include <hip/hip_runtime.h>
#include <hip/hip_bf16.h>

typedef __attribute__((ext_vector_type(16))) __bf16 v16bf;
typedef __attribute__((ext_vector_type(8)))  float  v8f;
typedef __attribute__((ext_vector_type(8)))  unsigned int v8u;

#define NUM_CODES 512
#define DIM 64
#define TOKENS_PER_WG 128   // 8 waves x 16 tokens

// LDS layout (dynamic):
//   [0,      65536)  : embedding as bf16, code-major, 32 uints (64 bf16) per code
//   [65536,  67584)  : float ||e||^2 per code (512 floats)
//   [67584,  68096)  : int argmin index per token tile (128 ints)
#define SMEM_BYTES (65536 + 2048 + 512)

__global__ __launch_bounds__(256) void vq_nearest_kernel(
    const float* __restrict__ inp,   // [32, 64, 64, 64] NCHW
    const float* __restrict__ emb,   // [512, 64]
    float* __restrict__ out)         // [32, 64, 64, 64] NCHW
{
  extern __shared__ __align__(16) char smem[];
  unsigned int* lds_emb  = (unsigned int*)smem;
  float*        lds_norm = (float*)(smem + 65536);
  int*          lds_idx  = (int*)(smem + 65536 + 2048);

  const int tid    = threadIdx.x;
  const int lane   = tid & 31;
  const int waveId = tid >> 5;
  const int half   = lane >> 4;   // which 16-lane half of the wave
  const int m      = lane & 15;   // row (token) / col (code) within tile

  // ---------------- Phase 0: load A fragments (tokens x features, bf16) ---
  // Token n -> (b, hw) in NCHW; feature K lives at ((b*64+K)<<12) + hw.
  // A 16x32, contiguous k order chosen here: lanes 0-15 hold K in {0..15},
  // lanes 16-31 hold K in {16..31} (rows M = lane&15).
  const int n  = blockIdx.x * TOKENS_PER_WG + waveId * 16 + m;
  const int b  = n >> 12;     // 4096 tokens per image
  const int hw = n & 4095;
  v16bf a0, a1;
#pragma unroll
  for (int i = 0; i < 16; ++i) {
    const int K = i + 16 * half;
    const float x0 = inp[((b * 64 + K) << 12) + hw];        // K in [0,32)
    const float x1 = inp[((b * 64 + K + 32) << 12) + hw];   // K in [32,64)
    a0[i] = (__bf16)x0;
    a1[i] = (__bf16)x1;
  }

  // ---------------- Phase 1: stage embedding -> LDS bf16 + norms ---------
  for (int c = tid; c < NUM_CODES; c += 256) {
    const float2* e2 = (const float2*)(emb + c * DIM);
    float nrm = 0.f;
#pragma unroll
    for (int j = 0; j < 32; ++j) {
      const float2 v = e2[j];
      nrm += v.x * v.x + v.y * v.y;
      const unsigned int lo = (unsigned int)__builtin_bit_cast(unsigned short, (__bf16)v.x);
      const unsigned int hi = (unsigned int)__builtin_bit_cast(unsigned short, (__bf16)v.y);
      lds_emb[c * 32 + j] = lo | (hi << 16);
    }
    lds_norm[c] = nrm;
  }
  __syncthreads();

  // ---------------- Phase 2: WMMA distances + running argmin -------------
  // C/D 16x16 f32 layout: VGPR v, lanes 0-15 -> (M=v, N=lane);
  // lanes 16-31 -> (M=v+8, N=lane-16). So per lane: 8 token rows, 1 code col.
  float bestVal[8];
  int   bestIdx[8];
#pragma unroll
  for (int v = 0; v < 8; ++v) { bestVal[v] = 3.4e38f; bestIdx[v] = 0; }

  for (int cb = 0; cb < NUM_CODES; cb += 16) {
    const int code = cb + m;   // this lane's code column
    // B 32x16, same contiguous k order as A: lane holds column N=code, K run of 16
    // starting at 16*half (frag0) / 32+16*half (frag1) -> exactly our LDS rows.
    const uint4* q = (const uint4*)(lds_emb + code * 32 + half * 8);
    const uint4 q0 = q[0], q1 = q[1], q2 = q[4], q3 = q[5];
    v8u r0, r1;
    r0[0]=q0.x; r0[1]=q0.y; r0[2]=q0.z; r0[3]=q0.w;
    r0[4]=q1.x; r0[5]=q1.y; r0[6]=q1.z; r0[7]=q1.w;
    r1[0]=q2.x; r1[1]=q2.y; r1[2]=q2.z; r1[3]=q2.w;
    r1[4]=q3.x; r1[5]=q3.y; r1[6]=q3.z; r1[7]=q3.w;
    const v16bf b0 = __builtin_bit_cast(v16bf, r0);
    const v16bf b1 = __builtin_bit_cast(v16bf, r1);

    v8f acc = {};
    acc = __builtin_amdgcn_wmma_f32_16x16x32_bf16(false, a0, false, b0,
                                                  (short)0, acc, false, false);
    acc = __builtin_amdgcn_wmma_f32_16x16x32_bf16(false, a1, false, b1,
                                                  (short)0, acc, false, false);

    const float nrm = lds_norm[code];
#pragma unroll
    for (int v = 0; v < 8; ++v) {
      const float dist = nrm - 2.0f * acc[v];   // ||x||^2 dropped (argmin-invariant)
      if (dist < bestVal[v]) { bestVal[v] = dist; bestIdx[v] = code; }
    }
  }

  // ---------------- Phase 3: argmin reduce across the 16-lane group ------
#pragma unroll
  for (int off = 1; off < 16; off <<= 1) {
#pragma unroll
    for (int v = 0; v < 8; ++v) {
      const float oV = __shfl_xor(bestVal[v], off, 32);
      const int   oI = __shfl_xor(bestIdx[v], off, 32);
      if (oV < bestVal[v] || (oV == bestVal[v] && oI < bestIdx[v])) {
        bestVal[v] = oV; bestIdx[v] = oI;
      }
    }
  }
  if (m == 0) {   // lane 0 has tokens M=0..7, lane 16 has tokens M=8..15
#pragma unroll
    for (int v = 0; v < 8; ++v)
      lds_idx[waveId * 16 + half * 8 + v] = bestIdx[v];
  }
  __syncthreads();

  // ---------------- Phase 4: gather code rows, scatter to NCHW -----------
  // 128 consecutive tokens share one contiguous HW span per channel, so for a
  // fixed d the 128 stores are 512B contiguous.
  const int tloc  = tid & 127;
  const int dBase = (tid >> 7) * 32;
  const int idx   = lds_idx[tloc];
  const int n2  = blockIdx.x * TOKENS_PER_WG + tloc;
  const int b2  = n2 >> 12;
  const int hw2 = n2 & 4095;
#pragma unroll
  for (int i = 0; i < 32; ++i) {
    const int d = dBase + i;
    out[((b2 * 64 + d) << 12) + hw2] = emb[idx * 64 + d];
  }
}

extern "C" void kernel_launch(void* const* d_in, const int* in_sizes, int n_in,
                              void* d_out, int out_size, void* d_ws, size_t ws_size,
                              hipStream_t stream) {
  (void)in_sizes; (void)n_in; (void)out_size; (void)d_ws; (void)ws_size;
  const float* inp = (const float*)d_in[0];   // [32,64,64,64] fp32
  const float* emb = (const float*)d_in[1];   // [512,64] fp32
  float* out = (float*)d_out;                 // [32,64,64,64] fp32

  const int total_tokens = 32 * 64 * 64;          // 131072
  const int blocks = total_tokens / TOKENS_PER_WG; // 1024
  vq_nearest_kernel<<<blocks, 256, SMEM_BYTES, stream>>>(inp, emb, out);
}
